// RotaryMultiHeadDotProductAttention_5592047419393
// MI455X (gfx1250) — matrix operand order repaired and checked by reading, not compiled, not run on hardware
//
#include <hip/hip_runtime.h>
#include <hip/hip_bf16.h>

typedef __bf16 bf16_t;
typedef __attribute__((ext_vector_type(8)))  bf16_t bf16x8;
typedef __attribute__((ext_vector_type(16))) bf16_t bf16x16;
typedef __attribute__((ext_vector_type(8)))  float  f32x8;
typedef __attribute__((ext_vector_type(4)))  unsigned int u32x4;
typedef __attribute__((ext_vector_type(8)))  int    i32x8;
typedef __attribute__((ext_vector_type(4)))  int    i32x4;

#define NHEADS 12
#define SEQ    1024
#define DMODEL 768
#define DHEAD  64
#define NBATCH 16

#if defined(__has_builtin)
#if __has_builtin(__builtin_amdgcn_tensor_load_to_lds) && \
    __has_builtin(__builtin_amdgcn_s_wait_tensorcnt)
#define HAVE_TDM 1
#endif
#endif
#ifndef HAVE_TDM
#define HAVE_TDM 0
#endif

#define WMMA_BF16(a, b, c)                                                     \
  __builtin_amdgcn_wmma_f32_16x16x32_bf16(false, (a), false, (b), (short)0,    \
                                          (c), false, false)

// A-fragment (16x32, MxK) from row-major memory, per CDNA5 ISA layout:
// lane m = lane&15, K chunks [k0,k0+8) and [k0+16,k0+24), k0 = (lane>>4)*8.
static __device__ __forceinline__ bf16x16 load_a_frag(const bf16_t* base,
                                                      int stride) {
  const int lane = threadIdx.x & 31;
  const int m = lane & 15;
  const int k0 = (lane >> 4) << 3;
  const bf16_t* p = base + (size_t)m * stride + k0;
  bf16x8 lo = *reinterpret_cast<const bf16x8*>(p);
  bf16x8 hi = *reinterpret_cast<const bf16x8*>(p + 16);
  bf16x16 r;
#pragma unroll
  for (int i = 0; i < 8; ++i) { r[i] = lo[i]; r[i + 8] = hi[i]; }
  return r;
}

// B-fragment (32x16, KxN) loaded from B^T stored row-major [N][K]:
// lane n = lane&15, K chunks [k0,k0+8) and [k0+16,k0+24), k0 = (lane>>4)*8.
static __device__ __forceinline__ bf16x16 load_b_frag(const bf16_t* baseT,
                                                      int stride) {
  const int lane = threadIdx.x & 31;
  const int n = lane & 15;
  const int k0 = (lane >> 4) << 3;
  const bf16_t* p = baseT + (size_t)n * stride + k0;
  bf16x8 lo = *reinterpret_cast<const bf16x8*>(p);
  bf16x8 hi = *reinterpret_cast<const bf16x8*>(p + 16);
  bf16x16 r;
#pragma unroll
  for (int i = 0; i < 8; ++i) { r[i] = lo[i]; r[i + 8] = hi[i]; }
  return r;
}

__global__ void cast_f32_to_bf16_kernel(const float* __restrict__ src,
                                        bf16_t* __restrict__ dst, int n) {
  int i = blockIdx.x * blockDim.x + threadIdx.x;
  if (i < n) dst[i] = (bf16_t)src[i];
}

// Transpose 768x768 weight matrices to [out][in] bf16 (B^T form for WMMA).
__global__ void prep_weights_kernel(const float* __restrict__ Wq,
                                    const float* __restrict__ Wk,
                                    const float* __restrict__ Wv,
                                    const float* __restrict__ Wo,
                                    bf16_t* __restrict__ WqT,
                                    bf16_t* __restrict__ WkT,
                                    bf16_t* __restrict__ WvT,
                                    bf16_t* __restrict__ WoT) {
  int i = blockIdx.x * blockDim.x + threadIdx.x;
  if (i >= DMODEL * DMODEL) return;
  int o = i / DMODEL, c = i - o * DMODEL;
  size_t src = (size_t)c * DMODEL + o;
  WqT[i] = (bf16_t)Wq[src];
  WkT[i] = (bf16_t)Wk[src];
  WvT[i] = (bf16_t)Wv[src];
  WoT[i] = (bf16_t)Wo[src];
}

// QKV projection: [16384 x 768] x [768 x 768] + bias, then RoPE (modes 0/1)
// or transposed store (mode 2: V^T [n][H][D][L]). One wave per 16x64 tile.
__global__ void qkv_proj_kernel(const bf16_t* __restrict__ X,
                                const bf16_t* __restrict__ WT,
                                const float* __restrict__ bias,
                                bf16_t* __restrict__ out, int mode) {
  const int m0 = blockIdx.x * 16;
  const int n0 = blockIdx.y * 64;
  const int lane = threadIdx.x & 31;
  const int ncol = lane & 15;
  const int rowbase = (lane >> 4) << 3;

  f32x8 acc0 = {}, acc1 = {}, acc2 = {}, acc3 = {};
  const bf16_t* xrow = X + (size_t)m0 * DMODEL;
  for (int kk = 0; kk < DMODEL; kk += 32) {
    bf16x16 a = load_a_frag(xrow + kk, DMODEL);
    bf16x16 b0 = load_b_frag(WT + (size_t)(n0) * DMODEL + kk, DMODEL);
    bf16x16 b1 = load_b_frag(WT + (size_t)(n0 + 16) * DMODEL + kk, DMODEL);
    bf16x16 b2 = load_b_frag(WT + (size_t)(n0 + 32) * DMODEL + kk, DMODEL);
    bf16x16 b3 = load_b_frag(WT + (size_t)(n0 + 48) * DMODEL + kk, DMODEL);
    acc0 = WMMA_BF16(a, b0, acc0);
    acc1 = WMMA_BF16(a, b1, acc1);
    acc2 = WMMA_BF16(a, b2, acc2);
    acc3 = WMMA_BF16(a, b3, acc3);
  }

  f32x8 accs[4] = {acc0, acc1, acc2, acc3};
#pragma unroll
  for (int j = 0; j < 4; ++j) {
    const int col = n0 + 16 * j + ncol;
    const int h = col >> 6;
    const int dd = col & 63;
    const float bv = bias[col];
    if (mode == 2) {
      // V^T store: lane holds 8 consecutive sequence positions at fixed dd.
      bf16x8 pk;
#pragma unroll
      for (int r = 0; r < 8; ++r) pk[r] = (bf16_t)(accs[j][r] + bv);
      const int nb = m0 >> 10;
      const int l0 = (m0 & (SEQ - 1)) + rowbase;
      *reinterpret_cast<bf16x8*>(
          out + ((((size_t)nb * NHEADS + h) * DHEAD + dd) * SEQ + l0)) = pk;
    } else {
#pragma unroll
      for (int r = 0; r < 8; ++r) {
        const int row = m0 + rowbase + r;
        const int l = row & (SEQ - 1);
        float val = accs[j][r] + bv;
        // 2D RoPE: pairs 0..15 use x-angle, 16..31 use y-angle; pair partner
        // (even/odd dd) lives in the adjacent lane of the C layout.
        const int pair = dd >> 1;
        const int p = (pair < 16) ? pair : (pair - 16);
        const float invf = __expf(-0.6140226914650779f * (float)p); // 1e4^(-p/15)
        const float pos = (pair < 16)
                              ? ((float)(l & 31) + 0.5f) * (1.0f / 32.0f)
                              : ((float)(l >> 5) + 0.5f) * (1.0f / 32.0f);
        const float ang = pos * invf;
        const float c = __cosf(ang), s = __sinf(ang);
        const float pv = __shfl_xor(val, 1, 32);
        val = (dd & 1) ? (pv * s + val * c) : (val * c - pv * s);
        const int nb = row >> 10;
        out[(((size_t)nb * NHEADS + h) * SEQ + l) * DHEAD + dd] = (bf16_t)val;
      }
    }
  }
}

// ---------------- attention with TDM double-buffered K/V staging ------------

// Per-wave dynamic-LDS layout (bytes): K tiles [2][4096], V tiles [2][4096],
// P relayout tile 16x40 bf16 (1280, padded to 1536).
#define ATT_KOFF(buf) ((unsigned)(buf) * 4096u)
#define ATT_VOFF(buf) (8192u + (unsigned)(buf) * 4096u)
#define ATT_POFF      16384u
#define ATT_WAVE_LDS  17920u
#define ATT_WAVES     2

#if HAVE_TDM
// Issue a TDM 2-D tile load (bf16 elements) global -> LDS.
// D# bitfields per CDNA5 ISA; groups 2/3 zero
// (tile_dim2..4 = 0 -> unused; 2-D descriptor). This toolchain declares the
// 6-arg builtin: (u32x4 g0, i32x8 g1, i32x4, i32x4, i32x8, i32 cpol).
static __device__ __forceinline__ void tdm_load_2d(unsigned lds_byte_addr,
                                                   const bf16_t* gptr,
                                                   unsigned tile_d0,
                                                   unsigned tile_d1,
                                                   unsigned long long stride0) {
  unsigned long long ga = (unsigned long long)(size_t)gptr;
  const unsigned td0 = 1u << 20, td1 = 1u << 20; // generous tensor dims
  u32x4 g0;
  g0[0] = 1u;                                   // count=1, user descriptor
  g0[1] = lds_byte_addr;                        // LDS byte address
  g0[2] = (unsigned)ga;                         // global_addr[31:0]
  g0[3] = (unsigned)((ga >> 32) & 0x01ffffffull) | (2u << 30); // type=2
  i32x8 g1;
  g1[0] = 0x00010000;                           // data_size = 2 bytes
  g1[1] = (int)((td0 & 0xffffu) << 16);         // tensor_dim0 lo
  g1[2] = (int)((td0 >> 16) | ((td1 & 0xffffu) << 16));
  g1[3] = (int)((td1 >> 16) | (tile_d0 << 16)); // tile_dim0
  g1[4] = (int)(tile_d1 & 0xffffu);             // tile_dim1 (tile_dim2 = 0)
  g1[5] = (int)(stride0 & 0xffffffffull);       // tensor_dim0_stride lo
  g1[6] = (int)((stride0 >> 32) & 0xffffull);   // stride hi (dim1_stride = 0)
  g1[7] = 0;
  i32x4 z4 = {0, 0, 0, 0};
  i32x8 z8 = {0, 0, 0, 0, 0, 0, 0, 0};
  __builtin_amdgcn_tensor_load_to_lds(g0, g1, z4, z4, z8, 0);
}
#define TDM_WAIT(n) __builtin_amdgcn_s_wait_tensorcnt(n)
#else
// Fallback: synchronous vectorized global -> LDS copy (per-lane 16B chunks).
static __device__ __forceinline__ void tdm_load_2d(unsigned lds_byte_addr,
                                                   const bf16_t* gptr,
                                                   unsigned tile_d0,
                                                   unsigned tile_d1,
                                                   unsigned long long stride0) {
  extern __shared__ char smem_fb[];
  const int lane = threadIdx.x & 31;
  const unsigned chunks = tile_d0 * tile_d1 / 8; // 8 bf16 per 16B chunk
  for (unsigned i = lane; i < chunks; i += 32) {
    unsigned row = (i * 8) / tile_d0;
    unsigned col = (i * 8) % tile_d0;
    *reinterpret_cast<u32x4*>(smem_fb + lds_byte_addr + (size_t)i * 16) =
        *reinterpret_cast<const u32x4*>(gptr + (size_t)row * stride0 + col);
  }
}
#define TDM_WAIT(n) ((void)0)
#endif

// Flash attention, one wave per (batch, head, 16-query tile).
// S^T = K*Q^T so each lane owns a query column -> in-lane online softmax.
// K and V^T chunk tiles are DMA'd into LDS by the Tensor Data Mover and
// double-buffered (TENSORcnt); O^T = V^T * P^T with P relayout through LDS.
__global__ void attention_kernel(const bf16_t* __restrict__ Q,
                                 const bf16_t* __restrict__ K,
                                 const bf16_t* __restrict__ Vt,
                                 bf16_t* __restrict__ AO) {
  extern __shared__ __align__(16) char smem[];
  const int wave = __builtin_amdgcn_readfirstlane((int)(threadIdx.x >> 5));
  const int lane = threadIdx.x & 31;
  const int ncol = lane & 15;
  const int rowbase = (lane >> 4) << 3;
  const unsigned wbase = (unsigned)wave * ATT_WAVE_LDS;
  char* wls = smem + wbase;

  const int tile = blockIdx.x * ATT_WAVES + wave;
  const int qt = tile & 63;
  const int h = (tile >> 6) % NHEADS;
  const int nb = tile / (64 * NHEADS);

  const size_t head = ((size_t)nb * NHEADS + h) * SEQ * DHEAD;
  const bf16_t* Qb = Q + head + (size_t)qt * 16 * DHEAD;
  const bf16_t* Kg = K + head;  // [SEQ][DHEAD]
  const bf16_t* Vg = Vt + head; // [DHEAD][SEQ]

  const bf16x16 bq0 = load_b_frag(Qb, DHEAD);      // d = 0..31
  const bf16x16 bq1 = load_b_frag(Qb + 32, DHEAD); // d = 32..63

  // Preload chunk 0: K chunk is 32x64 contiguous (one 4KB line-block),
  // V^T chunk is 64 rows x 32 keys strided by SEQ.
  tdm_load_2d(wbase + ATT_KOFF(0), Kg, 2048, 1, 2048);
  tdm_load_2d(wbase + ATT_VOFF(0), Vg, 32, 64, SEQ);

  f32x8 o0 = {}, o1 = {}, o2 = {}, o3 = {};
  float mrun = -1e30f, lrun = 0.0f;
  const float scale = 0.125f; // 1/sqrt(64)

  for (int c = 0; c < SEQ / 32; ++c) {
    const int kc = c * 32;
    const int buf = c & 1;
    if (c + 1 < SEQ / 32) {
      tdm_load_2d(wbase + ATT_KOFF(buf ^ 1), Kg + (size_t)(kc + 32) * DHEAD,
                  2048, 1, 2048);
      tdm_load_2d(wbase + ATT_VOFF(buf ^ 1), Vg + (kc + 32), 32, 64, SEQ);
      TDM_WAIT(2); // the two older (current-chunk) DMAs have landed
    } else {
      TDM_WAIT(0);
    }
    const bf16_t* Kl = (const bf16_t*)(wls + ATT_KOFF(buf)); // [32][64]
    const bf16_t* Vl = (const bf16_t*)(wls + ATT_VOFF(buf)); // [64][32]

    f32x8 s0 = {}, s1 = {};
    {
      bf16x16 a0 = load_a_frag(Kl, DHEAD);
      bf16x16 a1 = load_a_frag(Kl + 32, DHEAD);
      s0 = WMMA_BF16(a0, bq0, s0);
      s0 = WMMA_BF16(a1, bq1, s0);
    }
    {
      bf16x16 a0 = load_a_frag(Kl + 16 * DHEAD, DHEAD);
      bf16x16 a1 = load_a_frag(Kl + 16 * DHEAD + 32, DHEAD);
      s1 = WMMA_BF16(a0, bq0, s1);
      s1 = WMMA_BF16(a1, bq1, s1);
    }
    float cmax = -1e30f;
#pragma unroll
    for (int r = 0; r < 8; ++r) {
      s0[r] *= scale;
      s1[r] *= scale;
      cmax = fmaxf(cmax, fmaxf(s0[r], s1[r]));
    }
    cmax = fmaxf(cmax, __shfl_xor(cmax, 16, 32)); // other key half, same query
    const float mnew = fmaxf(mrun, cmax);
    const float corr = __expf(mrun - mnew);
    float csum = 0.0f;
    bf16x8 p0, p1;
#pragma unroll
    for (int r = 0; r < 8; ++r) {
      const float e0 = __expf(s0[r] - mnew);
      const float e1 = __expf(s1[r] - mnew);
      csum += e0 + e1;
      p0[r] = (bf16_t)e0;
      p1[r] = (bf16_t)e1;
    }
    csum += __shfl_xor(csum, 16, 32);
    lrun = lrun * corr + csum;
    mrun = mnew;
#pragma unroll
    for (int r = 0; r < 8; ++r) {
      o0[r] *= corr; o1[r] *= corr; o2[r] *= corr; o3[r] *= corr;
    }
    // C-layout (keys r+rowbase, query ncol) -> P^T B-fragment via LDS.
    // Single-wave tile; LDS ops are in-order within a wave on CDNA5.
    bf16_t* Pl = (bf16_t*)(wls + ATT_POFF); // [16][40]
    *reinterpret_cast<bf16x8*>(Pl + ncol * 40 + rowbase) = p0;
    *reinterpret_cast<bf16x8*>(Pl + ncol * 40 + rowbase + 16) = p1;
    const bf16x16 bp = load_b_frag(Pl, 40);
    o0 = WMMA_BF16(load_a_frag(Vl + 0 * 16 * 32, 32), bp, o0);
    o1 = WMMA_BF16(load_a_frag(Vl + 1 * 16 * 32, 32), bp, o1);
    o2 = WMMA_BF16(load_a_frag(Vl + 2 * 16 * 32, 32), bp, o2);
    o3 = WMMA_BF16(load_a_frag(Vl + 3 * 16 * 32, 32), bp, o3);
  }

  const float inv_l = 1.0f / lrun;
  const int q = qt * 16 + ncol;
  bf16_t* orow = AO + (((size_t)nb * SEQ + q) * NHEADS + h) * DHEAD;
  f32x8 os[4] = {o0, o1, o2, o3};
#pragma unroll
  for (int t = 0; t < 4; ++t) {
    bf16x8 pk;
#pragma unroll
    for (int r = 0; r < 8; ++r) pk[r] = (bf16_t)(os[t][r] * inv_l);
    *reinterpret_cast<bf16x8*>(orow + t * 16 + rowbase) = pk;
  }
}

// Output projection: [16384 x 768] x [768 x 768] + bo -> fp32 d_out.
__global__ void out_proj_kernel(const bf16_t* __restrict__ A,
                                const bf16_t* __restrict__ WT,
                                const float* __restrict__ bias,
                                float* __restrict__ out) {
  const int m0 = blockIdx.x * 16;
  const int n0 = blockIdx.y * 64;
  const int lane = threadIdx.x & 31;
  const int ncol = lane & 15;
  const int rowbase = (lane >> 4) << 3;

  f32x8 acc0 = {}, acc1 = {}, acc2 = {}, acc3 = {};
  const bf16_t* arow = A + (size_t)m0 * DMODEL;
  for (int kk = 0; kk < DMODEL; kk += 32) {
    bf16x16 a = load_a_frag(arow + kk, DMODEL);
    acc0 = WMMA_BF16(a, load_b_frag(WT + (size_t)(n0) * DMODEL + kk, DMODEL), acc0);
    acc1 = WMMA_BF16(a, load_b_frag(WT + (size_t)(n0 + 16) * DMODEL + kk, DMODEL), acc1);
    acc2 = WMMA_BF16(a, load_b_frag(WT + (size_t)(n0 + 32) * DMODEL + kk, DMODEL), acc2);
    acc3 = WMMA_BF16(a, load_b_frag(WT + (size_t)(n0 + 48) * DMODEL + kk, DMODEL), acc3);
  }
  f32x8 accs[4] = {acc0, acc1, acc2, acc3};
#pragma unroll
  for (int j = 0; j < 4; ++j) {
    const int col = n0 + 16 * j + ncol;
    const float bv = bias[col];
#pragma unroll
    for (int r = 0; r < 8; ++r) {
      out[(size_t)(m0 + rowbase + r) * DMODEL + col] = accs[j][r] + bv;
    }
  }
}

extern "C" void kernel_launch(void* const* d_in, const int* in_sizes, int n_in,
                              void* d_out, int out_size, void* d_ws,
                              size_t ws_size, hipStream_t stream) {
  (void)in_sizes; (void)n_in; (void)out_size; (void)ws_size;
  const float* inq = (const float*)d_in[0];
  const float* inkv = (const float*)d_in[1];
  const float* Wq = (const float*)d_in[2];
  const float* bq = (const float*)d_in[3];
  const float* Wk = (const float*)d_in[4];
  const float* bk = (const float*)d_in[5];
  const float* Wv = (const float*)d_in[6];
  const float* bv = (const float*)d_in[7];
  const float* Wo = (const float*)d_in[8];
  const float* bo = (const float*)d_in[9];
  float* outp = (float*)d_out;

  const size_t ACT = (size_t)NBATCH * SEQ * DMODEL * sizeof(bf16_t); // 24 MiB
  const size_t WSZ = (size_t)DMODEL * DMODEL * sizeof(bf16_t);       // 1.125 MiB
  char* ws = (char*)d_ws;
  bf16_t* Xq  = (bf16_t*)ws; ws += ACT;
  bf16_t* Xkv = (bf16_t*)ws; ws += ACT;
  bf16_t* WqT = (bf16_t*)ws; ws += WSZ;
  bf16_t* WkT = (bf16_t*)ws; ws += WSZ;
  bf16_t* WvT = (bf16_t*)ws; ws += WSZ;
  bf16_t* WoT = (bf16_t*)ws; ws += WSZ;
  bf16_t* Qb  = (bf16_t*)ws; ws += ACT; // [n][H][L][D]
  bf16_t* Kb  = (bf16_t*)ws; ws += ACT; // [n][H][L][D]
  bf16_t* Vtb = (bf16_t*)ws; ws += ACT; // [n][H][D][L]
  bf16_t* AO  = (bf16_t*)ws; ws += ACT; // [n][L][H][D]

  const int total = NBATCH * SEQ * DMODEL;
  cast_f32_to_bf16_kernel<<<(total + 255) / 256, 256, 0, stream>>>(inq, Xq, total);
  cast_f32_to_bf16_kernel<<<(total + 255) / 256, 256, 0, stream>>>(inkv, Xkv, total);
  prep_weights_kernel<<<(DMODEL * DMODEL + 255) / 256, 256, 0, stream>>>(
      Wq, Wk, Wv, Wo, WqT, WkT, WvT, WoT);

  dim3 gproj(NBATCH * SEQ / 16, DMODEL / 64);
  qkv_proj_kernel<<<gproj, 32, 0, stream>>>(Xq, WqT, bq, Qb, 0);
  qkv_proj_kernel<<<gproj, 32, 0, stream>>>(Xkv, WkT, bk, Kb, 1);
  qkv_proj_kernel<<<gproj, 32, 0, stream>>>(Xkv, WvT, bv, Vtb, 2);

  const int ntiles = NBATCH * NHEADS * (SEQ / 16);
  attention_kernel<<<ntiles / ATT_WAVES, 32 * ATT_WAVES,
                     ATT_WAVES * ATT_WAVE_LDS, stream>>>(Qb, Kb, Vtb, AO);

  out_proj_kernel<<<gproj, 32, 0, stream>>>(AO, WoT, bo, outp);
}
